// AttentionLSTMDecoderV1_53154515256104
// MI455X (gfx1250) — compile-verified, matrix operand order checked by reading, not run on hardware
//
#include <hip/hip_runtime.h>
#include <hip/hip_bf16.h>

// ---- problem constants --------------------------------------------------
#define BB   16
#define TT   500
#define DD   512
#define NN   128
#define EE   640
#define HH   1024
#define AAx  1024
#define PPx  1024
#define VV   10025
#define VPAD 10032      // 627 tiles of 16
#define KIH  1152       // E + D
#define G4H  4096       // 4*H
#define CATW 2176       // H + E + D

typedef __attribute__((ext_vector_type(16))) __bf16 v16bf;
typedef __attribute__((ext_vector_type(8)))  __bf16 v8bf;
typedef __attribute__((ext_vector_type(8)))  float  v8f;

__device__ __forceinline__ float sigmoidf_(float x) { return 1.f / (1.f + __expf(-x)); }

// Load one 16x32 bf16 A/B fragment from a row-major [rows x K] matrix.
// ISA layout (CDNA5 ISA, 16-bit A 16x32): lane L(0..15) = row L, K in
// {kh*8+0..7, 16+kh*8+0..7} with kh = lane>>4.  Both K runs are contiguous,
// so two 16B loads per lane (compiler fuses them into global_load_b128).
// B fragments from W[Nout,K] row-major use identical addressing (lane = col).
__device__ __forceinline__ v16bf load_frag_bf16(const __bf16* __restrict__ base,
                                                int ld, int r0, int k0, int lane) {
  int m  = lane & 15;
  int kh = (lane >> 4) & 1;
  const __bf16* p = base + (size_t)(r0 + m) * ld + k0 + kh * 8;
  v8bf lo = *(const v8bf*)p;
  v8bf hi = *(const v8bf*)(p + 16);
  return __builtin_shufflevector(lo, hi, 0,1,2,3,4,5,6,7,8,9,10,11,12,13,14,15);
}

// Generic bf16 WMMA GEMM: out[M,Ntot] = A[M,K] @ W[Nout,K]^T (+bias)
// 4 waves / block, each wave owns one 16x16 tile; blockIdx.y = M tile.
__global__ void wmma_gemm_bf16_kernel(const __bf16* __restrict__ A, int lda,
                                      const __bf16* __restrict__ W, int ldw,
                                      void* __restrict__ out, int ldo,
                                      const float* __restrict__ bias,
                                      int K, int ntiles, int Ntot, int out_is_bf16) {
  int lane = threadIdx.x & 31;
  int wv   = threadIdx.x >> 5;
  int nt   = blockIdx.x * 4 + wv;
  if (nt >= ntiles) return;                    // wave-uniform: EXEC stays all-1s
  int m0 = blockIdx.y * 16;
  int n0 = nt * 16;
  v8f acc = {0.f,0.f,0.f,0.f,0.f,0.f,0.f,0.f};
  for (int k0 = 0; k0 < K; k0 += 32) {
    v16bf af = load_frag_bf16(A, lda, m0, k0, lane);
    v16bf bf = load_frag_bf16(W, ldw, n0, k0, lane);
    acc = __builtin_amdgcn_wmma_f32_16x16x32_bf16(false, af, false, bf,
                                                  (short)0, acc, false, false);
  }
  // C/D layout: VGPR r, lanes0-15 -> M=r, lanes16-31 -> M=8+r; N = lane&15
  int col = n0 + (lane & 15);
  if (col >= Ntot) return;
  int rb = m0 + ((lane >> 4) & 1) * 8;
  float bv = bias ? bias[col] : 0.f;
  if (out_is_bf16) {
    __bf16* o = (__bf16*)out;
#pragma unroll
    for (int r = 0; r < 8; ++r) o[(size_t)(rb + r) * ldo + col] = (__bf16)(acc[r] + bv);
  } else {
    float* o = (float*)out;
#pragma unroll
    for (int r = 0; r < 8; ++r) o[(size_t)(rb + r) * ldo + col] = acc[r] + bv;
  }
}

// ---- helpers ------------------------------------------------------------
__global__ void cvt_f32_bf16_kernel(const float* __restrict__ s, __bf16* __restrict__ d, long n) {
  long i = (long)blockIdx.x * blockDim.x + threadIdx.x;
  long st = (long)gridDim.x * blockDim.x;
  for (; i < n; i += st) d[i] = (__bf16)s[i];
}

__global__ void zero_f32_kernel(float* __restrict__ d, long n) {
  long i = (long)blockIdx.x * blockDim.x + threadIdx.x;
  long st = (long)gridDim.x * blockDim.x;
  for (; i < n; i += st) d[i] = 0.f;
}

// W_out [V, P/2] -> bf16 padded to VPAD rows (zero fill beyond vocab)
__global__ void cvt_wout_pad_kernel(const float* __restrict__ s, __bf16* __restrict__ d) {
  long i = (long)blockIdx.x * blockDim.x + threadIdx.x;
  long n = (long)VPAD * (PPx / 2);
  long st = (long)gridDim.x * blockDim.x;
  for (; i < n; i += st) {
    int row = (int)(i / (PPx / 2));
    d[i] = (row < VV) ? (__bf16)s[i] : (__bf16)0.f;
  }
}

__global__ void bias_sum_kernel(const float* a, const float* b, float* o, int n) {
  int i = blockIdx.x * blockDim.x + threadIdx.x;
  if (i < n) o[i] = a[i] + b[i];
}

// shifted embedding gather -> cat[b,n, H .. H+E) (bf16)
__global__ void embed_kernel(const float* __restrict__ embed, const int* __restrict__ labels,
                             __bf16* __restrict__ cat) {
  long i = (long)blockIdx.x * blockDim.x + threadIdx.x;
  if (i >= (long)BB * NN * EE) return;
  int e = (int)(i % EE);
  int n = (int)((i / EE) % NN);
  int b = (int)(i / ((long)EE * NN));
  float v = 0.f;
  if (n > 0) {
    int tok = labels[b * NN + n - 1];
    v = embed[(size_t)tok * EE + e];
  }
  cat[((size_t)(b * NN + n)) * CATW + HH + e] = (__bf16)v;
}

// inv_fert[b,t] = sigmoid(enc[b,t,:] . W_if)   one wave per (b,t)
__global__ void invfert_kernel(const float* __restrict__ enc, const float* __restrict__ Wif,
                               float* __restrict__ invf) {
  int g    = (int)((blockIdx.x * blockDim.x + threadIdx.x) >> 5);
  int lane = threadIdx.x & 31;
  if (g >= BB * TT) return;
  const float* row = enc + (size_t)g * DD;
  float s = 0.f;
  for (int d = lane; d < DD; d += 32) s += row[d] * Wif[d];
#pragma unroll
  for (int off = 16; off; off >>= 1) s += __shfl_xor(s, off, 32);
  if (lane == 0) invf[g] = sigmoidf_(s);
}

// per-step: xcat[b, 0..E) = emb(step), xcat[b, E..E+D) = att_ctx
__global__ void build_xcat_kernel(const __bf16* __restrict__ cat, const float* __restrict__ attctx,
                                  __bf16* __restrict__ xcat, int step) {
  int i = blockIdx.x * blockDim.x + threadIdx.x;
  if (i >= BB * KIH) return;
  int b = i / KIH, c = i % KIH;
  __bf16 v;
  if (c < EE) v = cat[((size_t)(b * NN + step)) * CATW + HH + c];
  else        v = (__bf16)attctx[b * DD + (c - EE)];
  xcat[i] = v;
}

// zero-state LSTM cell (gate order i,f,g,o) + zoneout blend
__global__ void lstm_act_kernel(const float* __restrict__ gates,
                                float* __restrict__ hprev, float* __restrict__ cprev,
                                __bf16* __restrict__ hbf, __bf16* __restrict__ cat, int step) {
  int i = blockIdx.x * blockDim.x + threadIdx.x;
  if (i >= BB * HH) return;
  int b = i >> 10, hh = i & (HH - 1);
  const float* g = gates + (size_t)b * G4H;
  float gi = g[hh], gg = g[2 * HH + hh], go = g[3 * HH + hh];
  float c_new = sigmoidf_(gi) * tanhf(gg);
  float h_new = sigmoidf_(go) * tanhf(c_new);
  float h = 0.05f * hprev[i] + 0.95f * h_new;   // ZH = 0.05
  float c = 0.15f * cprev[i] + 0.85f * c_new;   // ZC = 0.15
  hprev[i] = h; cprev[i] = c;
  hbf[i] = (__bf16)h;
  cat[((size_t)(b * NN + step)) * CATW + hh] = (__bf16)h;
}

// additive attention + softmax + context + fertility accum. 1 block / batch.
// Energy loop: one wave per frame t; each lane owns 8 consecutive 'a' values
// -> one global_load_b128 of bf16 enc_ctx (L2-resident) and 16B LDS reads.
__global__ void __launch_bounds__(512)
attention_kernel(const __bf16* __restrict__ encctx, const float* __restrict__ enc,
                 const float* __restrict__ s_t, const float* __restrict__ Wfb,
                 const float* __restrict__ vatt, const float* __restrict__ invf,
                 const int* __restrict__ seqlen, float* __restrict__ accum,
                 float* __restrict__ attctx, __bf16* __restrict__ cat, int step) {
  __shared__ __align__(16) float s_sh[AAx], fb_sh[AAx], v_sh[AAx];
  __shared__ float en[512];
  __shared__ float red[512];
  int b = blockIdx.x, tid = threadIdx.x;
  for (int a = tid; a < AAx; a += 512) {
    s_sh[a]  = s_t[b * AAx + a];
    fb_sh[a] = Wfb[a];
    v_sh[a]  = vatt[a];
  }
  __syncthreads();
  int lane = tid & 31, wv = tid >> 5;
  int len = seqlen[b];
  for (int t = wv; t < TT; t += 16) {
    float ac = accum[b * TT + t];
    const __bf16* er = encctx + ((size_t)(b * TT + t)) * AAx;
    float s = 0.f;
#pragma unroll
    for (int a0 = 0; a0 < AAx; a0 += 256) {
      int base = a0 + lane * 8;
      v8bf e8 = *(const v8bf*)(er + base);
      float4 sa = *(const float4*)(s_sh + base);
      float4 sb = *(const float4*)(s_sh + base + 4);
      float4 fa = *(const float4*)(fb_sh + base);
      float4 fb = *(const float4*)(fb_sh + base + 4);
      float4 va = *(const float4*)(v_sh + base);
      float4 vb = *(const float4*)(v_sh + base + 4);
      s += tanhf((float)e8[0] + sa.x + ac * fa.x) * va.x;
      s += tanhf((float)e8[1] + sa.y + ac * fa.y) * va.y;
      s += tanhf((float)e8[2] + sa.z + ac * fa.z) * va.z;
      s += tanhf((float)e8[3] + sa.w + ac * fa.w) * va.w;
      s += tanhf((float)e8[4] + sb.x + ac * fb.x) * vb.x;
      s += tanhf((float)e8[5] + sb.y + ac * fb.y) * vb.y;
      s += tanhf((float)e8[6] + sb.z + ac * fb.z) * vb.z;
      s += tanhf((float)e8[7] + sb.w + ac * fb.w) * vb.w;
    }
#pragma unroll
    for (int off = 16; off; off >>= 1) s += __shfl_xor(s, off, 32);
    if (lane == 0) en[t] = (t < len) ? s : -3.0e38f;
  }
  if (tid >= TT) en[tid] = -3.0e38f;
  __syncthreads();
  // softmax over T
  float v = en[tid];
  red[tid] = v;
  __syncthreads();
  for (int s = 256; s; s >>= 1) { if (tid < s) red[tid] = fmaxf(red[tid], red[tid + s]); __syncthreads(); }
  float mx = red[0];
  __syncthreads();
  float ex = (tid < TT) ? __expf(v - mx) : 0.f;
  red[tid] = ex;
  __syncthreads();
  for (int s = 256; s; s >>= 1) { if (tid < s) red[tid] += red[tid + s]; __syncthreads(); }
  float w = ex / red[0];
  en[tid] = (tid < TT) ? w : 0.f;
  __syncthreads();
  // context: ctx[d] = sum_t w[t]*enc[b,t,d]  (fp32 enc is L2-resident,
  // contiguous d across threads -> coalesced 128B/wave)
  float cs = 0.f;
  int d = tid;  // D == 512 == blockDim
  for (int t = 0; t < TT; ++t) cs += en[t] * enc[((size_t)(b * TT + t)) * DD + d];
  attctx[b * DD + d] = cs;
  cat[((size_t)(b * NN + step)) * CATW + HH + EE + d] = (__bf16)cs;
  if (tid < TT) accum[b * TT + tid] += en[tid] * invf[b * TT + tid] * 0.5f;
}

// MaxOut(2) over readout, emit bf16 for the vocab GEMM
__global__ void maxout_kernel(const float* __restrict__ ro, __bf16* __restrict__ rd) {
  long i = (long)blockIdx.x * blockDim.x + threadIdx.x;
  if (i >= (long)BB * NN * (PPx / 2)) return;
  int row = (int)(i >> 9), c = (int)(i & 511);
  float v = fmaxf(ro[(size_t)row * PPx + 2 * c], ro[(size_t)row * PPx + 2 * c + 1]);
  rd[i] = (__bf16)v;
}

// ---- host side ----------------------------------------------------------
extern "C" void kernel_launch(void* const* d_in, const int* in_sizes, int n_in,
                              void* d_out, int out_size, void* d_ws, size_t ws_size,
                              hipStream_t stream) {
  const float* enc    = (const float*)d_in[0];
  const int*   labels = (const int*)d_in[1];
  const int*   seqlen = (const int*)d_in[2];
  const float* embed  = (const float*)d_in[3];
  const float* W_ih   = (const float*)d_in[4];
  const float* b_ih   = (const float*)d_in[5];
  const float* b_hh   = (const float*)d_in[6];
  const float* W_s    = (const float*)d_in[7];
  const float* W_ec   = (const float*)d_in[8];
  const float* b_ec   = (const float*)d_in[9];
  const float* v_att  = (const float*)d_in[10];
  const float* W_if   = (const float*)d_in[11];
  const float* W_fb   = (const float*)d_in[12];
  const float* W_ro   = (const float*)d_in[13];
  const float* b_ro   = (const float*)d_in[14];
  const float* W_out  = (const float*)d_in[15];
  const float* b_out  = (const float*)d_in[16];

  // ---- carve workspace (~72 MB) ----
  char* p = (char*)d_ws;
  auto alloc = [&](size_t bytes) -> void* {
    void* r = (void*)p; p += (bytes + 255) & ~(size_t)255; return r;
  };
  __bf16* enc_bf   = (__bf16*)alloc((size_t)BB * TT * DD * 2);
  __bf16* Wih_bf   = (__bf16*)alloc((size_t)G4H * KIH * 2);
  __bf16* Ws_bf    = (__bf16*)alloc((size_t)AAx * HH * 2);
  __bf16* Wec_bf   = (__bf16*)alloc((size_t)AAx * DD * 2);
  __bf16* Wro_bf   = (__bf16*)alloc((size_t)PPx * CATW * 2);
  __bf16* Wout_bf  = (__bf16*)alloc((size_t)VPAD * (PPx / 2) * 2);
  __bf16* encctx_bf= (__bf16*)alloc((size_t)BB * TT * AAx * 2);
  float*  invf     = (float*)alloc((size_t)BB * TT * 4);
  float*  accum    = (float*)alloc((size_t)BB * TT * 4);
  __bf16* cat      = (__bf16*)alloc((size_t)BB * NN * CATW * 2);
  float*  readout  = (float*)alloc((size_t)BB * NN * PPx * 4);
  __bf16* rd_bf    = (__bf16*)alloc((size_t)BB * NN * (PPx / 2) * 2);
  float*  bsum     = (float*)alloc((size_t)G4H * 4);
  float*  hprev    = (float*)alloc((size_t)BB * HH * 4);
  float*  cprev    = (float*)alloc((size_t)BB * HH * 4);
  __bf16* hbf      = (__bf16*)alloc((size_t)BB * HH * 2);
  float*  gates    = (float*)alloc((size_t)BB * G4H * 4);
  float*  s_t      = (float*)alloc((size_t)BB * AAx * 4);
  float*  attctx   = (float*)alloc((size_t)BB * DD * 4);
  __bf16* xcat     = (__bf16*)alloc((size_t)BB * KIH * 2);

  // ---- weight / input conversion to bf16 ----
  cvt_f32_bf16_kernel<<<2048, 256, 0, stream>>>(enc,  enc_bf, (long)BB * TT * DD);
  cvt_f32_bf16_kernel<<<2048, 256, 0, stream>>>(W_ih, Wih_bf, (long)G4H * KIH);
  cvt_f32_bf16_kernel<<<1024, 256, 0, stream>>>(W_s,  Ws_bf,  (long)AAx * HH);
  cvt_f32_bf16_kernel<<<1024, 256, 0, stream>>>(W_ec, Wec_bf, (long)AAx * DD);
  cvt_f32_bf16_kernel<<<2048, 256, 0, stream>>>(W_ro, Wro_bf, (long)PPx * CATW);
  cvt_wout_pad_kernel<<<2048, 256, 0, stream>>>(W_out, Wout_bf);
  bias_sum_kernel<<<(G4H + 255) / 256, 256, 0, stream>>>(b_ih, b_hh, bsum, G4H);

  // ---- zero state ----
  zero_f32_kernel<<<64, 256, 0, stream>>>(accum,  (long)BB * TT);
  zero_f32_kernel<<<64, 256, 0, stream>>>(attctx, (long)BB * DD);
  zero_f32_kernel<<<64, 256, 0, stream>>>(hprev,  (long)BB * HH);
  zero_f32_kernel<<<64, 256, 0, stream>>>(cprev,  (long)BB * HH);

  // ---- shifted embeddings into cat[:, H:H+E) ----
  embed_kernel<<<((long)BB * NN * EE + 255) / 256, 256, 0, stream>>>(embed, labels, cat);

  // ---- enc_ctx = enc @ W_ec^T + b_ec  (bf16 out, L2-resident) ----
  {
    dim3 g(AAx / 16 / 4, (BB * TT) / 16);     // (16, 500)
    wmma_gemm_bf16_kernel<<<g, 128, 0, stream>>>(enc_bf, DD, Wec_bf, DD,
                                                 encctx_bf, AAx, b_ec,
                                                 DD, AAx / 16, AAx, 1);
  }
  invfert_kernel<<<(BB * TT * 32 + 255) / 256, 256, 0, stream>>>(enc, W_if, invf);

  // ---- 128 sequential decoder steps ----
  for (int step = 0; step < NN; ++step) {
    build_xcat_kernel<<<(BB * KIH + 255) / 256, 256, 0, stream>>>(cat, attctx, xcat, step);
    {
      dim3 g(G4H / 16 / 4, 1);                // 256 tiles, M=16
      wmma_gemm_bf16_kernel<<<g, 128, 0, stream>>>(xcat, KIH, Wih_bf, KIH,
                                                   gates, G4H, bsum,
                                                   KIH, G4H / 16, G4H, 0);
    }
    lstm_act_kernel<<<(BB * HH + 255) / 256, 256, 0, stream>>>(gates, hprev, cprev, hbf, cat, step);
    {
      dim3 g(AAx / 16 / 4, 1);                // s_t = h @ W_s^T
      wmma_gemm_bf16_kernel<<<g, 128, 0, stream>>>(hbf, HH, Ws_bf, HH,
                                                   s_t, AAx, nullptr,
                                                   HH, AAx / 16, AAx, 0);
    }
    attention_kernel<<<BB, 512, 0, stream>>>(encctx_bf, enc, s_t, W_fb, v_att,
                                             invf, seqlen, accum, attctx, cat, step);
  }

  // ---- readout GEMM + MaxOut + logits GEMM ----
  {
    dim3 g(PPx / 16 / 4, (BB * NN) / 16);     // (16, 128)
    wmma_gemm_bf16_kernel<<<g, 128, 0, stream>>>(cat, CATW, Wro_bf, CATW,
                                                 readout, PPx, b_ro,
                                                 CATW, PPx / 16, PPx, 0);
  }
  maxout_kernel<<<((long)BB * NN * (PPx / 2) + 255) / 256, 256, 0, stream>>>(readout, rd_bf);
  {
    int ntiles = VPAD / 16;                   // 627
    dim3 g((ntiles + 3) / 4, (BB * NN) / 16); // (157, 128)
    wmma_gemm_bf16_kernel<<<g, 128, 0, stream>>>(rd_bf, PPx / 2, Wout_bf, PPx / 2,
                                                 (float*)d_out, VV, b_out,
                                                 PPx / 2, ntiles, VV, 0);
  }
}
